// mh_attn_50122268344420
// MI455X (gfx1250) — compile-verified, matrix operand order checked by reading, not run on hardware
//
#include <hip/hip_runtime.h>
#include <hip/hip_bf16.h>

// ---------------------------------------------------------------------------
// Multi-head attention (Swin-style: logit_scale + sigmoid rel-pos bias)
// B=4, S=2048, D=512, H=8, hd=64.  f16 WMMA (v_wmma_f32_16x16x32_f16),
// f32 accumulate, fused flash-attention middle stage.
// ---------------------------------------------------------------------------

#define BATCH 4
#define SEQ   2048
#define DMODEL 512
#define HEADS 8
#define HDIM  64

typedef __attribute__((ext_vector_type(16))) _Float16 v16h;
typedef __attribute__((ext_vector_type(8)))  float    v8f;

__device__ __forceinline__ v8f wmma_f16(v16h a, v16h b, v8f c) {
  // (neg_a, A, neg_b, B, c_mod, C, reuse_a, reuse_b)
  return __builtin_amdgcn_wmma_f32_16x16x32_f16(false, a, false, b, (short)0, c,
                                                false, false);
}

// A/B operand loader (16x32 f16 tile, one row per lane).
// p must already point at row_base + k_chunk + (lane>=16 ? 8 : 0).
// Elements [0..7] = K+0..7 of this lane's chunk, [8..15] = K+16..23.
__device__ __forceinline__ v16h load_tile_f32(const float* __restrict__ p) {
  v16h a;
#pragma unroll
  for (int t = 0; t < 8; ++t) {
    a[t]     = (_Float16)p[t];
    a[t + 8] = (_Float16)p[t + 16];
  }
  return a;
}
__device__ __forceinline__ v16h load_tile_f16(const _Float16* __restrict__ p) {
  v16h a;
#pragma unroll
  for (int t = 0; t < 8; ++t) {
    a[t]     = p[t];
    a[t + 8] = p[t + 16];
  }
  return a;
}

// ---------------------------------------------------------------------------
// Kernel 0: sigmoid(rpe) -> f32 table  (4095*8 elements, lives in L2)
// ---------------------------------------------------------------------------
__global__ void rpe_sigmoid_kernel(const float* __restrict__ rpe,
                                   float* __restrict__ sig, int n) {
  int i = blockIdx.x * blockDim.x + threadIdx.x;
  if (i < n) sig[i] = 1.0f / (1.0f + __expf(-rpe[i]));
}

// ---------------------------------------------------------------------------
// Kernel 1: fused QKV projection.  X (8192x512 f32) @ W^T + b -> f16.
// grid = (M/64, N/64, 3)  block = 128 (4 waves, each wave: 16x64 strip)
// Q,K stored [B,H,S,64]; V stored transposed [B,H,64,S].
// Head split per reference: feature n -> h = n % 8, d = n / 8.
// ---------------------------------------------------------------------------
__global__ __launch_bounds__(128) void qkv_proj_kernel(
    const float* __restrict__ X,
    const float* __restrict__ Wq, const float* __restrict__ bq,
    const float* __restrict__ Wk, const float* __restrict__ bk,
    const float* __restrict__ Wv, const float* __restrict__ bv,
    _Float16* __restrict__ qf, _Float16* __restrict__ kf,
    _Float16* __restrict__ vt) {
  const int lane = threadIdx.x & 31;
  const int wave = threadIdx.x >> 5;
  const int lr = lane & 15;   // row-in-tile / n-in-tile
  const int hs = lane >> 4;   // half select
  const int mode = blockIdx.z;

  const float* W    = (mode == 0) ? Wq : (mode == 1) ? Wk : Wv;
  const float* bias = (mode == 0) ? bq : (mode == 1) ? bk : bv;

  const int m0 = blockIdx.x * 64 + wave * 16;
  const int n0 = blockIdx.y * 64;

  v8f acc[4] = {{}, {}, {}, {}};
  const float* arow = X + (m0 + lr) * DMODEL + hs * 8;

  for (int kk = 0; kk < DMODEL; kk += 32) {
    v16h a = load_tile_f32(arow + kk);
#pragma unroll
    for (int sub = 0; sub < 4; ++sub) {
      const float* brow = W + (n0 + sub * 16 + lr) * DMODEL + kk + hs * 8;
      v16h bm = load_tile_f32(brow);
      acc[sub] = wmma_f16(a, bm, acc[sub]);
    }
  }

#pragma unroll
  for (int sub = 0; sub < 4; ++sub) {
    const int n = n0 + sub * 16 + lr;
    const int hh = n & 7, d = n >> 3;  // f = d*H + h  (reference split_heads)
    const float bn = bias[n];
#pragma unroll
    for (int r = 0; r < 8; ++r) {
      const int m = m0 + r + 8 * hs;
      const int bidx = m >> 11;      // / SEQ
      const int s = m & (SEQ - 1);
      const _Float16 val = (_Float16)(acc[sub][r] + bn);
      if (mode == 0)
        qf[((bidx * HEADS + hh) * SEQ + s) * HDIM + d] = val;
      else if (mode == 1)
        kf[((bidx * HEADS + hh) * SEQ + s) * HDIM + d] = val;
      else
        vt[((bidx * HEADS + hh) * HDIM + d) * SEQ + s] = val;  // transposed
    }
  }
}

// ---------------------------------------------------------------------------
// Kernel 2: flash attention.
// grid = (S/64, H, B), block = 128 (4 waves, each owns 16 query rows).
// Per 64-wide k-tile: 8 WMMA (QK^T), online softmax, P->LDS->A-layout,
// 8 WMMA (PV).  Output written f16 to ao [B,S,D] with f = h*64 + d.
// ---------------------------------------------------------------------------
__global__ __launch_bounds__(128) void flash_attn_kernel(
    const _Float16* __restrict__ qf, const _Float16* __restrict__ kf,
    const _Float16* __restrict__ vt, const float* __restrict__ sig_rpe,
    const float* __restrict__ logit_scale, _Float16* __restrict__ ao) {
  // padded to 66 halves/row: row stride 33 dwords, gcd(33,64)=1 -> no bank conflicts
  __shared__ _Float16 plds[4][16][66];

  const int lane = threadIdx.x & 31;
  const int wave = threadIdx.x >> 5;
  const int lr = lane & 15;
  const int hs = lane >> 4;

  const int b = blockIdx.z;
  const int h = blockIdx.y;
  const int q0 = blockIdx.x * 64 + wave * 16;

  // scale = exp(min(ls, ln(100))) / sqrt(D_MODEL)
  const float scale =
      __expf(fminf(logit_scale[h], 4.6051702f)) * 0.04419417382f;

  const int bh = b * HEADS + h;

  // Q strip: 16 rows x 64 (two K-chunks of 32)
  const _Float16* qp = qf + ((size_t)bh * SEQ + (q0 + lr)) * HDIM + hs * 8;
  v16h aq[2];
  aq[0] = load_tile_f16(qp);
  aq[1] = load_tile_f16(qp + 32);

  v8f o[4] = {{}, {}, {}, {}};
  float mrow[8], lrow[8];
#pragma unroll
  for (int r = 0; r < 8; ++r) { mrow[r] = -1e30f; lrow[r] = 0.0f; }

  for (int kt = 0; kt < SEQ / 64; ++kt) {
    const int j0 = kt * 64;

    // ---- scores: S = Q K^T * scale + sigmoid(rpe[i-j+S-1, h]) ----
    v8f sv[4];
#pragma unroll
    for (int sub = 0; sub < 4; ++sub) {
      const _Float16* kp =
          kf + ((size_t)bh * SEQ + (j0 + sub * 16 + lr)) * HDIM + hs * 8;
      v8f s = {};
      s = wmma_f16(aq[0], load_tile_f16(kp), s);
      s = wmma_f16(aq[1], load_tile_f16(kp + 32), s);
      const int j = j0 + sub * 16 + lr;  // this lane's column
#pragma unroll
      for (int r = 0; r < 8; ++r) {
        const int i = q0 + r + 8 * hs;   // this element's row
        const float bias = sig_rpe[(i - j + (SEQ - 1)) * HEADS + h];
        s[r] = s[r] * scale + bias;
      }
      sv[sub] = s;
    }

    // ---- tile row-max (reduce over 4 subs then across 16 lanes) ----
    float tmax[8];
#pragma unroll
    for (int r = 0; r < 8; ++r)
      tmax[r] = fmaxf(fmaxf(sv[0][r], sv[1][r]), fmaxf(sv[2][r], sv[3][r]));
#pragma unroll
    for (int msk = 1; msk < 16; msk <<= 1)
#pragma unroll
      for (int r = 0; r < 8; ++r)
        tmax[r] = fmaxf(tmax[r], __shfl_xor(tmax[r], msk, 32));

    // ---- online softmax update ----
    float corr[8];
#pragma unroll
    for (int r = 0; r < 8; ++r) {
      const float nm = fmaxf(mrow[r], tmax[r]);
      corr[r] = __expf(mrow[r] - nm);
      mrow[r] = nm;
    }
    float tsum[8] = {0, 0, 0, 0, 0, 0, 0, 0};
#pragma unroll
    for (int sub = 0; sub < 4; ++sub)
#pragma unroll
      for (int r = 0; r < 8; ++r) {
        const float p = __expf(sv[sub][r] - mrow[r]);
        sv[sub][r] = p;
        tsum[r] += p;
      }
#pragma unroll
    for (int msk = 1; msk < 16; msk <<= 1)
#pragma unroll
      for (int r = 0; r < 8; ++r) tsum[r] += __shfl_xor(tsum[r], msk, 32);
#pragma unroll
    for (int r = 0; r < 8; ++r) {
      lrow[r] = lrow[r] * corr[r] + tsum[r];
#pragma unroll
      for (int sd = 0; sd < 4; ++sd) o[sd][r] *= corr[r];
    }

    // ---- P: C-layout regs -> LDS (f16) -> A-layout regs ----
    __syncthreads();
#pragma unroll
    for (int sub = 0; sub < 4; ++sub)
#pragma unroll
      for (int r = 0; r < 8; ++r)
        plds[wave][r + 8 * hs][sub * 16 + lr] = (_Float16)sv[sub][r];
    __syncthreads();

    v16h ap[2];
#pragma unroll
    for (int c = 0; c < 2; ++c) {
      const int cb = c * 32 + hs * 8;
#pragma unroll
      for (int t = 0; t < 8; ++t) {
        ap[c][t]     = plds[wave][lr][cb + t];
        ap[c][t + 8] = plds[wave][lr][cb + t + 16];
      }
    }

    // ---- O += P @ V   (V^T rows are head-dims; contiguous over j) ----
#pragma unroll
    for (int sd = 0; sd < 4; ++sd) {
      const _Float16* vp =
          vt + ((size_t)bh * HDIM + sd * 16 + lr) * SEQ + j0 + hs * 8;
      o[sd] = wmma_f16(ap[0], load_tile_f16(vp), o[sd]);
      o[sd] = wmma_f16(ap[1], load_tile_f16(vp + 32), o[sd]);
    }
  }

  // ---- epilogue: O / l, merge heads with f = h*64 + d ----
#pragma unroll
  for (int sd = 0; sd < 4; ++sd) {
    const int d = sd * 16 + lr;
#pragma unroll
    for (int r = 0; r < 8; ++r) {
      const int i = q0 + r + 8 * hs;
      const float val = o[sd][r] / lrow[r];
      ao[((size_t)b * SEQ + i) * DMODEL + h * HDIM + d] = (_Float16)val;
    }
  }
}

// ---------------------------------------------------------------------------
// Kernel 3: output projection.  AO (8192x512 f16) @ Wp^T + bp -> f32 d_out.
// ---------------------------------------------------------------------------
__global__ __launch_bounds__(128) void out_proj_kernel(
    const _Float16* __restrict__ ao, const float* __restrict__ Wp,
    const float* __restrict__ bp, float* __restrict__ out) {
  const int lane = threadIdx.x & 31;
  const int wave = threadIdx.x >> 5;
  const int lr = lane & 15;
  const int hs = lane >> 4;

  const int m0 = blockIdx.x * 64 + wave * 16;
  const int n0 = blockIdx.y * 64;

  v8f acc[4] = {{}, {}, {}, {}};
  const _Float16* arow = ao + (size_t)(m0 + lr) * DMODEL + hs * 8;

  for (int kk = 0; kk < DMODEL; kk += 32) {
    v16h a = load_tile_f16(arow + kk);
#pragma unroll
    for (int sub = 0; sub < 4; ++sub) {
      const float* brow = Wp + (n0 + sub * 16 + lr) * DMODEL + kk + hs * 8;
      acc[sub] = wmma_f16(a, load_tile_f32(brow), acc[sub]);
    }
  }

#pragma unroll
  for (int sub = 0; sub < 4; ++sub) {
    const int n = n0 + sub * 16 + lr;
    const float bn = bp[n];
#pragma unroll
    for (int r = 0; r < 8; ++r) {
      const int m = m0 + r + 8 * hs;
      out[(size_t)m * DMODEL + n] = acc[sub][r] + bn;
    }
  }
}

// ---------------------------------------------------------------------------
extern "C" void kernel_launch(void* const* d_in, const int* in_sizes, int n_in,
                              void* d_out, int out_size, void* d_ws,
                              size_t ws_size, hipStream_t stream) {
  (void)in_sizes; (void)n_in; (void)out_size; (void)ws_size;

  const float* query = (const float*)d_in[0];
  const float* Wq = (const float*)d_in[1];
  const float* bq = (const float*)d_in[2];
  const float* Wk = (const float*)d_in[3];
  const float* bk = (const float*)d_in[4];
  const float* Wv = (const float*)d_in[5];
  const float* bv = (const float*)d_in[6];
  const float* Wp = (const float*)d_in[7];
  const float* bp = (const float*)d_in[8];
  const float* logit_scale = (const float*)d_in[9];
  const float* rpe = (const float*)d_in[10];

  // workspace carve-up
  const size_t QKV_ELEMS = (size_t)BATCH * HEADS * SEQ * HDIM;  // 4.19M halves
  char* ws = (char*)d_ws;
  float* sig = (float*)ws;                         // 4095*8 f32 = 131040 B
  _Float16* qf = (_Float16*)(ws + 131072);
  _Float16* kf = qf + QKV_ELEMS;
  _Float16* vt = kf + QKV_ELEMS;
  _Float16* ao = vt + QKV_ELEMS;

  const int nrpe = (2 * SEQ - 1) * HEADS;
  rpe_sigmoid_kernel<<<(nrpe + 255) / 256, 256, 0, stream>>>(rpe, sig, nrpe);

  qkv_proj_kernel<<<dim3((BATCH * SEQ) / 64, DMODEL / 64, 3), 128, 0, stream>>>(
      query, Wq, bq, Wk, bk, Wv, bv, qf, kf, vt);

  flash_attn_kernel<<<dim3(SEQ / 64, HEADS, BATCH), 128, 0, stream>>>(
      qf, kf, vt, sig, logit_scale, ao);

  out_proj_kernel<<<dim3((BATCH * SEQ) / 64, DMODEL / 64), 128, 0, stream>>>(
      ao, Wp, bp, (float*)d_out);
}
